// MultiHeadGraphAttention_54142357733801
// MI455X (gfx1250) — matrix operand order repaired and checked by reading, not compiled, not run on hardware
//
#include <hip/hip_runtime.h>
#include <hip/hip_bf16.h>

// ---------------------------------------------------------------------------
// Multi-head GAT for MI455X (gfx1250, wave32, WMMA).
//   H=8 heads, N=4096 nodes, F_IN=256, F_OUT=64.
// Pipeline:
//   K1: h_prime[h] = h @ w[h]            (bf16 WMMA, fp32 accum) -> ws (f32 + bf16)
//   K2: score[h,n] = log2e * h_prime[h,n,:]·a[h]                 -> ws
//   K3: flash-style  out = softmax(mask(lrelu(s_i+s_j))) @ h_prime
//       one 256-thread block per 16-row strip; wave w == head w so the
//       adjacency stream (64 MB = HBM roofline term) is read once per strip.
//       Softmax denominator comes from a 5th WMMA against an all-ones B tile
//       (moves reduction work from the saturated VALU to the idle XDL pipe,
//       and lands den[row] exactly in the lane/VGPR that writes that row).
// ---------------------------------------------------------------------------

typedef __attribute__((ext_vector_type(16))) __bf16 v16bf;
typedef __attribute__((ext_vector_type(8)))  float  v8f;
typedef __attribute__((ext_vector_type(4)))  float  f4;
typedef __attribute__((ext_vector_type(4)))  int    i4;
typedef __attribute__((ext_vector_type(16))) short  v16s;

#define N_NODES 4096
#define F_IN    256
#define F_OUT   64
#define LOG2E   1.4426950408889634f

// ---------------------------------------------------------------------------
// Kernel 1: h_prime = h @ w  per head.  One wave computes a 16x64 strip.
// A (16x32 bf16) per ISA layout: lane<16 -> M=lane, K = {0..7,16..23};
//                                lane>=16 -> M=lane-16, K = {8..15,24..31}.
// B (32x16 bf16): lane -> N=lane&15, elements -> K same order as A.
// D (16x16 f32):  lane<16 -> N=lane, M=v; lane>=16 -> N=lane-16, M=v+8.
// ---------------------------------------------------------------------------
__global__ __launch_bounds__(256) void gat_hprime_kernel(
    const float* __restrict__ h, const float* __restrict__ w,
    float* __restrict__ hp_f32, __bf16* __restrict__ hp_bf16)
{
  const int lane = threadIdx.x & 31;
  const int wid  = (blockIdx.x * blockDim.x + threadIdx.x) >> 5; // 0..2047
  const int head = wid >> 8;            // 8 heads x 256 strips
  const int m0   = (wid & 255) << 4;    // 16-row strip

  const int m  = lane & 15;
  const int kb = (lane < 16) ? 0 : 8;

  const float* hrow  = h + (size_t)(m0 + m) * F_IN;
  const float* wbase = w + (size_t)head * F_IN * F_OUT;

  v8f acc[4];
#pragma unroll
  for (int t = 0; t < 4; ++t) acc[t] = (v8f){0.f,0.f,0.f,0.f,0.f,0.f,0.f,0.f};

  for (int k0 = 0; k0 < F_IN; k0 += 32) {
    // ---- stage ALL loads first so they can clause / overlap ----
    f4 alo0 = *(const f4*)(hrow + k0 + kb);
    f4 alo1 = *(const f4*)(hrow + k0 + kb + 4);
    f4 ahi0 = *(const f4*)(hrow + k0 + kb + 16);
    f4 ahi1 = *(const f4*)(hrow + k0 + kb + 20);

    const float* wcol = wbase + (size_t)(k0 + kb) * F_OUT + m; // lane -> N
    float bb[64];
#pragma unroll
    for (int t = 0; t < 4; ++t) {
#pragma unroll
      for (int s = 0; s < 8; ++s) {
        bb[16*t+s]   = wcol[(size_t)s * F_OUT + 16*t];
        bb[16*t+8+s] = wcol[(size_t)(16 + s) * F_OUT + 16*t];
      }
    }

    v16bf a;
    a[0]=(__bf16)alo0.x; a[1]=(__bf16)alo0.y; a[2]=(__bf16)alo0.z; a[3]=(__bf16)alo0.w;
    a[4]=(__bf16)alo1.x; a[5]=(__bf16)alo1.y; a[6]=(__bf16)alo1.z; a[7]=(__bf16)alo1.w;
    a[8]=(__bf16)ahi0.x; a[9]=(__bf16)ahi0.y; a[10]=(__bf16)ahi0.z; a[11]=(__bf16)ahi0.w;
    a[12]=(__bf16)ahi1.x; a[13]=(__bf16)ahi1.y; a[14]=(__bf16)ahi1.z; a[15]=(__bf16)ahi1.w;

#pragma unroll
    for (int t = 0; t < 4; ++t) {
      v16bf b;
#pragma unroll
      for (int q = 0; q < 16; ++q) b[q] = (__bf16)bb[16*t+q];
      acc[t] = __builtin_amdgcn_wmma_f32_16x16x32_bf16(
          false, a, false, b, (short)0, acc[t], false, false);
    }
  }

  const int hi = lane >> 4;
#pragma unroll
  for (int t = 0; t < 4; ++t) {
    const int n = 16*t + (lane & 15);
#pragma unroll
    for (int v = 0; v < 8; ++v) {
      const int mrow = m0 + v + 8*hi;
      const size_t idx = ((size_t)head * N_NODES + mrow) * F_OUT + n;
      const float val = acc[t][v];
      hp_f32[idx]  = val;
      hp_bf16[idx] = (__bf16)val;
    }
  }
}

// ---------------------------------------------------------------------------
// Kernel 2: score[h,n] = log2(e) * dot(h_prime[h,n,:], a_src[h,:])
// Pre-scaling by log2(e) lets K3 use raw v_exp_f32 (exp2) with no per-element
// multiply: exp(lrelu(si+sj)) == exp2(lrelu(si'+sj')) since lrelu commutes
// with a positive scale.
// ---------------------------------------------------------------------------
__global__ __launch_bounds__(256) void gat_score_kernel(
    const float* __restrict__ hp_f32, const float* __restrict__ a_src,
    float* __restrict__ score)
{
  const int t = blockIdx.x * blockDim.x + threadIdx.x; // 0..32767
  const int head = t >> 12;
  const int n    = t & (N_NODES - 1);
  const float* row = hp_f32 + ((size_t)head * N_NODES + n) * F_OUT;
  const float* av  = a_src + head * F_OUT;
  float s = 0.f;
#pragma unroll
  for (int o = 0; o < F_OUT; o += 4) {
    f4 rv = *(const f4*)(row + o);
    f4 a4 = *(const f4*)(av + o);
    s += rv.x*a4.x + rv.y*a4.y + rv.z*a4.z + rv.w*a4.w;
  }
  score[t] = s * LOG2E;
}

// ---------------------------------------------------------------------------
// Kernel 3: streamed masked-softmax-matmul.
// Block = 256 thr = 8 waves; wave w handles head w over rows [i0, i0+16).
// P tile (16x32) built in registers in WMMA A layout; masked entries are 0,
// no online max needed (scores are O(10), exp2 range safe in f32/bf16).
// ---------------------------------------------------------------------------
#define GAT_P(Qv, Cv, IDX) do {                       \
    float x_ = s_row + (Cv);                          \
    x_ = fmaxf(x_, 0.2f * x_);   /* LeakyReLU(0.2) */ \
    float p_ = (Qv) ? __builtin_amdgcn_exp2f(x_) : 0.0f; \
    a[IDX] = (__bf16)p_;                              \
  } while (0)

__global__ __launch_bounds__(256) void gat_attn_kernel(
    const int* __restrict__ adj, const float* __restrict__ score,
    const __bf16* __restrict__ hp_bf16, const float* __restrict__ bias,
    float* __restrict__ out)
{
  const int lane = threadIdx.x & 31;
  const int head = threadIdx.x >> 5;   // wave id == head
  const int i0   = blockIdx.x << 4;    // 16-row strip

  const int m  = lane & 15;
  const int kb = (lane < 16) ? 0 : 8;

  const float  s_row  = score[head * N_NODES + i0 + m];
  const int*   adjrow = adj + (size_t)(i0 + m) * N_NODES;
  const float* scol   = score + head * N_NODES;
  const __bf16* hpb   = hp_bf16 + (size_t)head * N_NODES * F_OUT;

  // all-ones B tile: P @ ones gives softmax denominator via the XDL pipe
  v16bf bone;
#pragma unroll
  for (int q = 0; q < 16; ++q) bone[q] = (__bf16)1.0f;

  v8f acc[4];
#pragma unroll
  for (int t = 0; t < 4; ++t) acc[t] = (v8f){0.f,0.f,0.f,0.f,0.f,0.f,0.f,0.f};
  v8f accd = (v8f){0.f,0.f,0.f,0.f,0.f,0.f,0.f,0.f};

  for (int j0 = 0; j0 < N_NODES; j0 += 32) {
    // adjacency + column scores for this lane's 16 K-slots
    i4 qlo0 = *(const i4*)(adjrow + j0 + kb);
    i4 qlo1 = *(const i4*)(adjrow + j0 + kb + 4);
    i4 qhi0 = *(const i4*)(adjrow + j0 + kb + 16);
    i4 qhi1 = *(const i4*)(adjrow + j0 + kb + 20);
    f4 clo0 = *(const f4*)(scol + j0 + kb);
    f4 clo1 = *(const f4*)(scol + j0 + kb + 4);
    f4 chi0 = *(const f4*)(scol + j0 + kb + 16);
    f4 chi1 = *(const f4*)(scol + j0 + kb + 20);

    // prefetch next adj tile into cache (global_prefetch_b8)
    __builtin_prefetch(adjrow + j0 + 32 + kb, 0, 1);

    v16bf a;
    GAT_P(qlo0.x, clo0.x, 0);  GAT_P(qlo0.y, clo0.y, 1);
    GAT_P(qlo0.z, clo0.z, 2);  GAT_P(qlo0.w, clo0.w, 3);
    GAT_P(qlo1.x, clo1.x, 4);  GAT_P(qlo1.y, clo1.y, 5);
    GAT_P(qlo1.z, clo1.z, 6);  GAT_P(qlo1.w, clo1.w, 7);
    GAT_P(qhi0.x, chi0.x, 8);  GAT_P(qhi0.y, chi0.y, 9);
    GAT_P(qhi0.z, chi0.z, 10); GAT_P(qhi0.w, chi0.w, 11);
    GAT_P(qhi1.x, chi1.x, 12); GAT_P(qhi1.y, chi1.y, 13);
    GAT_P(qhi1.z, chi1.z, 14); GAT_P(qhi1.w, chi1.w, 15);

    const __bf16* bcol = hpb + (size_t)(j0 + kb) * F_OUT + m; // lane -> N
#pragma unroll
    for (int t = 0; t < 4; ++t) {
      v16bf b;
#pragma unroll
      for (int s = 0; s < 8; ++s) {
        b[s]   = bcol[(size_t)s * F_OUT + 16*t];
        b[8+s] = bcol[(size_t)(16 + s) * F_OUT + 16*t];
      }
      acc[t] = __builtin_amdgcn_wmma_f32_16x16x32_bf16(
          false, a, false, b, (short)0, acc[t], false, false);
    }
    // denominator: row-sum of P, replicated over N; lands per-lane in the
    // exact (row = v + 8*hi) slots the epilogue writes -> no shuffles needed
    accd = __builtin_amdgcn_wmma_f32_16x16x32_bf16(
        false, a, false, bone, (short)0, accd, false, false);
  }

  const int hi = lane >> 4;
  float dinv[8];
#pragma unroll
  for (int v = 0; v < 8; ++v) dinv[v] = 1.0f / accd[v];

#pragma unroll
  for (int t = 0; t < 4; ++t) {
    const int n  = 16*t + (lane & 15);
    const float bn = bias[n];
#pragma unroll
    for (int v = 0; v < 8; ++v) {
      float o = acc[t][v] * dinv[v] + bn;
      o = fmaxf(o, 0.01f * o);               // LeakyReLU(0.01)
      out[((size_t)head * N_NODES + (i0 + v + 8*hi)) * F_OUT + n] = o;
    }
  }
}

// ---------------------------------------------------------------------------
extern "C" void kernel_launch(void* const* d_in, const int* in_sizes, int n_in,
                              void* d_out, int out_size, void* d_ws, size_t ws_size,
                              hipStream_t stream) {
  const float* h     = (const float*)d_in[0];
  const int*   adj   = (const int*)  d_in[1];
  const float* w     = (const float*)d_in[2];
  const float* a_src = (const float*)d_in[3];
  const float* bias  = (const float*)d_in[4];
  float* out = (float*)d_out;

  char* ws = (char*)d_ws;
  float*  hp_f32  = (float*) (ws);                              // 8 MB
  __bf16* hp_bf16 = (__bf16*)(ws + (size_t)8  * 1024 * 1024);   // 4 MB
  float*  score   = (float*) (ws + (size_t)12 * 1024 * 1024);   // 128 KB

  // K1: h_prime (2048 waves = 8 heads x 256 strips)
  hipLaunchKernelGGL(gat_hprime_kernel, dim3(256), dim3(256), 0, stream,
                     h, w, hp_f32, hp_bf16);
  // K2: scores (32768 threads), pre-scaled by log2(e)
  hipLaunchKernelGGL(gat_score_kernel, dim3(128), dim3(256), 0, stream,
                     hp_f32, a_src, score);
  // K3: masked softmax @ h_prime, fused (256 row-blocks x 8 head-waves)
  hipLaunchKernelGGL(gat_attn_kernel, dim3(256), dim3(256), 0, stream,
                     adj, score, hp_bf16, bias, out);
}
